// Conv3DMod_62010737819927
// MI455X (gfx1250) — matrix operand order repaired and checked by reading, not compiled, not run on hardware
//
#include <hip/hip_runtime.h>

typedef __attribute__((ext_vector_type(16))) __bf16 v16bf;
typedef __attribute__((ext_vector_type(8)))  __bf16 v8bf;
typedef __attribute__((ext_vector_type(8)))  float  v8f;

#define EPS    1e-8f
#define C_IN   128
#define C_OUT  128
#define SDIM   32
#define NB     4
#define TAPS   27
#define LDIC   136   // padded LDS row stride (elements); 272B = 17*16B aligned
#define DROWS  34    // d' = -1..32

// ---------------- kernel 1: demod[b][oc] = rsqrt(sum((w*y)^2)+eps) ----------
__global__ void k_demod(const float* __restrict__ w, const float* __restrict__ y,
                        float* __restrict__ demod) {
  __shared__ float red[C_IN];
  int boc = blockIdx.x;                 // b*128 + oc
  int b = boc >> 7, oc = boc & 127;
  int ic = threadIdx.x;                 // 128 threads
  float s = y[b * C_IN + ic];
  const float* wp = w + ((size_t)oc * C_IN + ic) * TAPS;
  float acc = 0.f;
#pragma unroll
  for (int t = 0; t < TAPS; ++t) { float v = wp[t] * s; acc += v * v; }
  red[ic] = acc;
  __syncthreads();
  for (int off = 64; off > 0; off >>= 1) {
    if (ic < off) red[ic] += red[ic + off];
    __syncthreads();
  }
  if (ic == 0) demod[boc] = rsqrtf(red[0] + EPS);
}

// ------------- kernel 2: wmod[b][tap][oc][ic] = bf16(w*y*demod) -------------
__global__ void k_modw(const float* __restrict__ w, const float* __restrict__ y,
                       const float* __restrict__ demod, __bf16* __restrict__ wmod) {
  int idx = blockIdx.x * 256 + threadIdx.x;   // exact: 4*27*128*128 = 6912*256
  int ic  = idx & 127;
  int oc  = (idx >> 7) & 127;
  int rem = idx >> 14;
  int tap = rem % TAPS;
  int b   = rem / TAPS;
  float v = w[((size_t)oc * C_IN + ic) * TAPS + tap]
          * y[b * C_IN + ic] * demod[b * C_OUT + oc];
  wmod[idx] = (__bf16)v;
}

// ------------- kernel 3: implicit-GEMM conv via bf16 WMMA -------------------
// grid: b*32*32 workgroups (one per (b,h,w) output pencil of 128oc x 32d)
// block: 256 threads = 8 waves; wave => one 16-oc stripe, two 16x16 N-tiles
__global__ __launch_bounds__(256)
void k_conv(const float* __restrict__ x, const __bf16* __restrict__ wmod,
            float* __restrict__ out) {
  __shared__ __bf16 xs[DROWS * LDIC];   // 9.0 KB

  int wg = blockIdx.x;                  // b*1024 + h*32 + w
  int b  = wg >> 10;
  int h  = (wg >> 5) & 31;
  int w  = wg & 31;

  int tid  = threadIdx.x;
  int lane = tid & 31;
  int wave = tid >> 5;                  // 0..7
  int oc0  = wave * 16;

  v8f acc0 = {};                        // d 0..15
  v8f acc1 = {};                        // d 16..31

  int arow  = oc0 + (lane & 15);        // A: M row (oc)
  int ahalf = (lane >> 4);              // A: K-subgroup select
  int bcol  = lane & 15;                // B: N (d within half)
  int bks   = (lane >> 4) * 16;        // B: K-offset within 32-chunk

  const __bf16* wb = wmod + (size_t)b * TAPS * C_OUT * C_IN;

  for (int dh = 0; dh < 3; ++dh) {
    int hh = h + dh - 1;
    for (int dw = 0; dw < 3; ++dw) {
      int ww = w + dw - 1;
      if (hh < 0 || hh >= SDIM || ww < 0 || ww >= SDIM) continue;  // block-uniform

      __syncthreads();  // previous iteration's LDS reads complete
      // stage x[b][ic][hh][ww][d'] -> xs[d'+1][ic] as bf16 (transpose to ic-contig)
      const float* xb = x + (((size_t)b * C_IN) * SDIM + hh) * SDIM * SDIM
                          + (size_t)ww * SDIM;
      for (int idx = tid; idx < DROWS * C_IN; idx += 256) {
        int dzi = idx % DROWS;          // 0..33  (d = dzi-1)
        int ic  = idx / DROWS;
        int d   = dzi - 1;
        float v = 0.f;
        if (d >= 0 && d < SDIM)
          v = xb[(size_t)ic * (SDIM * SDIM * SDIM) + d];
        xs[dzi * LDIC + ic] = (__bf16)v;
      }
      __syncthreads();

      int tapbase = (dh * 3 + dw) * 3;
#pragma unroll
      for (int kz = 0; kz < 3; ++kz) {
        const __bf16* wt = wb + (size_t)(tapbase + kz) * C_OUT * C_IN
                              + (size_t)arow * C_IN;
#pragma unroll
        for (int kc = 0; kc < 4; ++kc) {
          int ic0 = kc * 32;
          // A operand: 16x32 bf16 tile of wmod, two aligned 16B loads
          v8bf alo = *(const v8bf*)(wt + ic0 + ahalf * 16);
          v8bf ahi = *(const v8bf*)(wt + ic0 + ahalf * 16 + 8);
          v16bf A = __builtin_shufflevector(alo, ahi,
                      0,1,2,3,4,5,6,7,8,9,10,11,12,13,14,15);
          // B operands: 32x16 bf16 tiles of staged x, contiguous LDS reads
          const __bf16* bp0 = xs + (bcol + kz) * LDIC + ic0 + bks;
          const __bf16* bp1 = xs + (16 + bcol + kz) * LDIC + ic0 + bks;
          v8bf b0l = *(const v8bf*)(bp0);
          v8bf b0h = *(const v8bf*)(bp0 + 8);
          v8bf b1l = *(const v8bf*)(bp1);
          v8bf b1h = *(const v8bf*)(bp1 + 8);
          v16bf B0 = __builtin_shufflevector(b0l, b0h,
                      0,1,2,3,4,5,6,7,8,9,10,11,12,13,14,15);
          v16bf B1 = __builtin_shufflevector(b1l, b1h,
                      0,1,2,3,4,5,6,7,8,9,10,11,12,13,14,15);
          acc0 = __builtin_amdgcn_wmma_f32_16x16x32_bf16(
                   false, A, false, B0, (short)0, acc0, false, false);
          acc1 = __builtin_amdgcn_wmma_f32_16x16x32_bf16(
                   false, A, false, B1, (short)0, acc1, false, false);
        }
      }
    }
  }

  // writeback: C/D layout -> (oc = r + 8*(lane/16), d = lane%16)
  int ocw  = oc0 + (lane >> 4) * 8;
  int dbas = lane & 15;
  float* op = out + ((((size_t)b * C_OUT + ocw) * SDIM + h) * SDIM + w) * SDIM;
#pragma unroll
  for (int r = 0; r < 8; ++r) {
    op[(size_t)r * (SDIM * SDIM * SDIM) + dbas]      = acc0[r];
    op[(size_t)r * (SDIM * SDIM * SDIM) + 16 + dbas] = acc1[r];
  }
}

extern "C" void kernel_launch(void* const* d_in, const int* in_sizes, int n_in,
                              void* d_out, int out_size, void* d_ws, size_t ws_size,
                              hipStream_t stream) {
  const float* x = (const float*)d_in[0];
  const float* y = (const float*)d_in[1];
  const float* w = (const float*)d_in[2];
  float* out = (float*)d_out;

  float*  demod = (float*)d_ws;                       // 512 floats
  __bf16* wmod  = (__bf16*)((char*)d_ws + 4096);      // 4*27*128*128 bf16 = 3.4MB

  k_demod<<<NB * C_OUT, C_IN, 0, stream>>>(w, y, demod);
  k_modw<<<(NB * TAPS * C_OUT * C_IN) / 256, 256, 0, stream>>>(w, y, demod, wmod);
  k_conv<<<NB * SDIM * SDIM, 256, 0, stream>>>(x, wmod, out);
}
